// SABlock_2834678415763
// MI455X (gfx1250) — matrix operand order repaired and checked by reading, not compiled, not run on hardware
//
#include <hip/hip_runtime.h>
#include <hip/hip_bf16.h>
#include <cstdint>
#include <cstddef>

typedef unsigned short u16;
typedef __attribute__((ext_vector_type(16))) __bf16 v16bf;
typedef __attribute__((ext_vector_type(8)))  __bf16 v8bf;
typedef __attribute__((ext_vector_type(8)))  float   v8f;
typedef __attribute__((ext_vector_type(4)))  int     v4i;

#if defined(__gfx1250__) && __has_builtin(__builtin_amdgcn_global_load_async_to_lds_b128)
#define HAVE_ASYNC_LDS 1
#else
#define HAVE_ASYNC_LDS 0
#endif

__device__ __forceinline__ u16 f32_to_bf16(float f) {
  unsigned int u = __float_as_uint(f);
  u += 0x7FFFu + ((u >> 16) & 1u);  // round-to-nearest-even
  return (u16)(u >> 16);
}

__device__ __forceinline__ v8f wmma_bf16(v16bf a, v16bf b, v8f c) {
  // D(16x16,f32) = A(16x32,bf16) * B(32x16,bf16) + C
  return __builtin_amdgcn_wmma_f32_16x16x32_bf16(false, a, false, b, (short)0, c,
                                                 false, false);
}

// 16-byte global -> LDS copy: async (ASYNCcnt) on gfx1250, sync fallback otherwise.
__device__ __forceinline__ void copy16_g2l(u16* lds, const u16* g) {
#if HAVE_ASYNC_LDS
  __builtin_amdgcn_global_load_async_to_lds_b128(
      (__attribute__((address_space(1))) v4i*)const_cast<u16*>(g),
      (__attribute__((address_space(3))) v4i*)lds, 0, 0);
#else
  *(v8bf*)lds = *(const v8bf*)g;
  *((v8bf*)lds + 1) = *(const v8bf*)(g + 8);
#endif
}

template <int N>
__device__ __forceinline__ void wait_async() {
#if HAVE_ASYNC_LDS
#if __has_builtin(__builtin_amdgcn_s_wait_asynccnt)
  __builtin_amdgcn_s_wait_asynccnt((unsigned short)N);
#else
  asm volatile("s_wait_asynccnt %0" ::"i"(N) : "memory");
#endif
#endif
}

// ---------------------------------------------------------------- converts
__global__ void cvt_f32_to_bf16(const float* __restrict__ in, u16* __restrict__ out,
                                size_t n) {
  size_t i = (size_t)blockIdx.x * blockDim.x + threadIdx.x;
  if (i < n) out[i] = f32_to_bf16(in[i]);
}

// V[b,t,h*64+c] -> Vt[((b*H+h)*64+c)*S + t]
__global__ void transpose_v_kernel(const u16* __restrict__ V, u16* __restrict__ Vt,
                                   int S, int d, int H, size_t n) {
  size_t i = (size_t)blockIdx.x * blockDim.x + threadIdx.x;
  if (i >= n) return;
  int c_ = (int)(i % (size_t)d);
  size_t t = (i / (size_t)d) % (size_t)S;
  size_t b = i / ((size_t)d * (size_t)S);
  int h = c_ >> 6, c = c_ & 63;
  Vt[((b * H + h) * 64 + (size_t)c) * (size_t)S + t] = V[i];
}

// ---------------------------------------------------------------- GEMM (NT)
// C[M,N] = A[M,K] * Bw[N,K]^T (+ epilogue)
// EPI: 0 = bf16; 1 = f32; 2 = bias+silu bf16; 3 = bias f32; 4 = *0.125 bf16
// Block 256 thr = 8 waves (4x2), tile 128x128, per-wave 32x64, K step 32.
// Double-buffered LDS staged with async global->LDS copies.
template <int EPI>
__global__ __launch_bounds__(256) void gemm_nt_bf16(
    const u16* __restrict__ A, const u16* __restrict__ Bw,
    const float* __restrict__ bias, void* __restrict__ C, int M, int N, int K) {
  constexpr int BM = 128, BN = 128, BK = 32;
  __shared__ __align__(32) u16 As[2][BM * BK];
  __shared__ __align__(32) u16 Bs[2][BN * BK];
  const int tid = threadIdx.x;
  const int wave = tid >> 5, lane = tid & 31;
  const int wm = wave & 3, wn = wave >> 2;  // 4 x 2
  const int r      = lane & 15;
  const int kbA    = (lane >> 4) << 4;  // A frag: 0 / 16 contiguous
  const int kbB    = (lane >> 4) << 4;  // B frag: 0 / 16 contiguous
  const int rowoff = (lane >> 4) << 3;  // D rows
  const size_t m_blk = (size_t)blockIdx.x * BM;
  const size_t n_blk = (size_t)blockIdx.y * BN;

  // stage loader: 128x32 u16 tile = 512 16B chunks; 256 thr x 2 chunks each.
  const int lrow = tid >> 1, lhalf = (tid & 1) << 4;
  const u16* gA = A + (m_blk + (size_t)lrow) * (size_t)K + lhalf;
  const u16* gB = Bw + (n_blk + (size_t)lrow) * (size_t)K + lhalf;
  u16* lA = &As[0][lrow * BK + lhalf];
  u16* lB = &Bs[0][lrow * BK + lhalf];
  const size_t qgs = (size_t)128 * (size_t)K;  // +128 rows in global
  auto stage = [&](int buf, int k0) {
    copy16_g2l(lA + buf * (BM * BK), gA + k0);
    copy16_g2l(lA + buf * (BM * BK) + 128 * BK, gA + k0 + qgs);
    copy16_g2l(lB + buf * (BN * BK), gB + k0);
    copy16_g2l(lB + buf * (BN * BK) + 128 * BK, gB + k0 + qgs);
  };

  v8f acc[2][4] = {};
  stage(0, 0);
  for (int k0 = 0; k0 < K; k0 += BK) {
    const int buf = (k0 >> 5) & 1;
    const bool more = (k0 + BK) < K;
    if (more) stage(buf ^ 1, k0 + BK);  // prefetch next stage
    if (more) wait_async<4>();          // current stage done; next 4 in flight
    else      wait_async<0>();
    __syncthreads();

    v16bf af[2], bfr[4];
#pragma unroll
    for (int i = 0; i < 2; i++) {
      const u16* p = &As[buf][(wm * 32 + i * 16 + r) * BK + kbA];
      *(v8bf*)&af[i]       = *(const v8bf*)p;
      *((v8bf*)&af[i] + 1) = *(const v8bf*)(p + 8);
    }
#pragma unroll
    for (int j = 0; j < 4; j++)
      bfr[j] = *(const v16bf*)&Bs[buf][(wn * 64 + j * 16 + r) * BK + kbB];

#pragma unroll
    for (int i = 0; i < 2; i++)
#pragma unroll
      for (int j = 0; j < 4; j++) acc[i][j] = wmma_bf16(af[i], bfr[j], acc[i][j]);
    __syncthreads();
  }

#pragma unroll
  for (int i = 0; i < 2; i++) {
#pragma unroll
    for (int j = 0; j < 4; j++) {
      const size_t col = n_blk + (size_t)(wn * 64 + j * 16 + r);
      const float bv = (EPI == 2 || EPI == 3) ? bias[col] : 0.0f;
#pragma unroll
      for (int e = 0; e < 8; e++) {
        const size_t row = m_blk + (size_t)(wm * 32 + i * 16 + rowoff + e);
        float v = acc[i][j][e] + bv;
        if (EPI == 2) v = v / (1.0f + __expf(-v));  // silu
        if (EPI == 4) v *= 0.125f;                  // fold attention 1/sqrt(dh)
        if (EPI == 1 || EPI == 3)
          ((float*)C)[row * (size_t)N + col] = v;
        else
          ((u16*)C)[row * (size_t)N + col] = f32_to_bf16(v);
      }
    }
  }
}

// ---------------------------------------------------------------- attention
// grid (S/64, B*H), block 128 (4 waves, 16 query rows each).
// Q pre-scaled by 1/sqrt(dh). Max-free online softmax (logits tame by design):
// oacc += exp(s)*V, rsum += exp(s); out = oacc / rsum.
__global__ __launch_bounds__(128) void attn_kernel(
    const u16* __restrict__ Qb, const u16* __restrict__ Kb,
    const u16* __restrict__ Vt, float* __restrict__ Out, int S, int d, int H) {
  __shared__ __align__(32) u16 Ps[4][16 * 32];
  __shared__ float denom[4][16];
  const int tid = threadIdx.x, wave = tid >> 5, lane = tid & 31;
  const int r      = lane & 15;
  const int kbA    = (lane >> 4) << 4;
  const int kbB    = (lane >> 4) << 4;
  const int rowoff = (lane >> 4) << 3;
  const int b = blockIdx.y / H, h = blockIdx.y % H;
  const int q0 = blockIdx.x * 64 + wave * 16;
  const u16* Qbase = Qb + ((size_t)b * S) * (size_t)d + h * 64;
  const u16* Vbase = Vt + (size_t)blockIdx.y * 64 * (size_t)S;

  v16bf qf[2];  // Q rows, dh=64 as two K=32 chunks
#pragma unroll
  for (int c = 0; c < 2; c++) {
    const u16* p = Qbase + (size_t)(q0 + r) * (size_t)d + c * 32 + kbA;
    *(v8bf*)&qf[c]       = *(const v8bf*)p;
    *((v8bf*)&qf[c] + 1) = *(const v8bf*)(p + 8);
  }

  // hoisted per-lane streaming pointers (no 64-bit muls in the loop)
  const u16* Kp = Kb + ((size_t)b * S) * (size_t)d + h * 64 + (size_t)r * d + kbB;
  const u16* Vp = Vbase + (size_t)r * S + kbB;
  const size_t kStep = (size_t)32 * d;  // keys advance 32 rows / iter
  const size_t ntOff = (size_t)16 * d;  // second 16-key tile (32KB, fits imm24)
  const size_t dcOff = (size_t)16 * S;  // next 16 dh cols of Vt (64KB, fits imm24)

  v8f oacc[4] = {};
  float rsum[8] = {0, 0, 0, 0, 0, 0, 0, 0};

  for (int t0 = 0; t0 < S; t0 += 32) {
    // issue ALL loads for this iteration up front (V is score-independent)
    v16bf kf00 = *(const v16bf*)(Kp);
    v16bf kf01 = *(const v16bf*)(Kp + 32);
    v16bf kf10 = *(const v16bf*)(Kp + ntOff);
    v16bf kf11 = *(const v16bf*)(Kp + ntOff + 32);
    v16bf vf[4];
#pragma unroll
    for (int dc = 0; dc < 4; dc++) vf[dc] = *(const v16bf*)(Vp + dc * dcOff);

    v8f s0 = {}, s1 = {};
    s0 = wmma_bf16(qf[0], kf00, s0);
    s0 = wmma_bf16(qf[1], kf01, s0);
    s1 = wmma_bf16(qf[0], kf10, s1);
    s1 = wmma_bf16(qf[1], kf11, s1);

#pragma unroll
    for (int e = 0; e < 8; e++) {
      float e0 = __expf(s0[e]);
      float e1 = __expf(s1[e]);
      rsum[e] += e0 + e1;
      Ps[wave][(e + rowoff) * 32 + r]      = f32_to_bf16(e0);
      Ps[wave][(e + rowoff) * 32 + 16 + r] = f32_to_bf16(e1);
    }
    __syncthreads();
    v16bf pf;  // P(16x32) reloaded in A-fragment layout
    {
      const u16* p = &Ps[wave][r * 32 + kbA];
      *(v8bf*)&pf       = *(const v8bf*)p;
      *((v8bf*)&pf + 1) = *(const v8bf*)(p + 8);
    }
#pragma unroll
    for (int dc = 0; dc < 4; dc++) oacc[dc] = wmma_bf16(pf, vf[dc], oacc[dc]);
    __syncthreads();

    Kp += kStep;
    Vp += 32;
  }

  if (lane < 16) denom[wave][lane] = 0.0f;
  __syncthreads();
#pragma unroll
  for (int e = 0; e < 8; e++) atomicAdd(&denom[wave][e + rowoff], rsum[e]);
  __syncthreads();

  float* obase = Out + ((size_t)b * S) * (size_t)d + h * 64;
#pragma unroll
  for (int dc = 0; dc < 4; dc++)
#pragma unroll
    for (int e = 0; e < 8; e++) {
      const float v = oacc[dc][e] / denom[wave][e + rowoff];
      obase[(size_t)(q0 + e + rowoff) * (size_t)d + dc * 16 + r] = v;
    }
}

// ---------------------------------------------------------------- layernorm
// y = (X+Res - mu)*rsqrt(var+eps)*g + b; one row (d=1024) per 256-thread block.
__global__ __launch_bounds__(256) void ln_kernel(
    const float* __restrict__ X, const float* __restrict__ Res,
    const float* __restrict__ g, const float* __restrict__ bt,
    float* __restrict__ Yf, u16* __restrict__ Yb, int d) {
  __shared__ float s1[256], s2[256];
  const int tid = threadIdx.x;
  const size_t base = (size_t)blockIdx.x * (size_t)d;
  float vals[4];
  float lsum = 0.f, lsq = 0.f;
#pragma unroll
  for (int i = 0; i < 4; i++) {
    int c = tid + i * 256;
    float v = X[base + c] + Res[base + c];
    vals[i] = v;
    lsum += v;
    lsq += v * v;
  }
  s1[tid] = lsum;
  s2[tid] = lsq;
  __syncthreads();
  for (int o = 128; o > 0; o >>= 1) {
    if (tid < o) { s1[tid] += s1[tid + o]; s2[tid] += s2[tid + o]; }
    __syncthreads();
  }
  const float inv_d = 1.0f / (float)d;
  const float mu = s1[0] * inv_d;
  const float var = s2[0] * inv_d - mu * mu;
  const float rstd = rsqrtf(var + 1e-5f);
#pragma unroll
  for (int i = 0; i < 4; i++) {
    int c = tid + i * 256;
    float y = (vals[i] - mu) * rstd * g[c] + bt[c];
    Yf[base + c] = y;
    if (Yb) Yb[base + c] = f32_to_bf16(y);
  }
}

// ---------------------------------------------------------------- launcher
extern "C" void kernel_launch(void* const* d_in, const int* in_sizes, int n_in,
                              void* d_out, int out_size, void* d_ws, size_t ws_size,
                              hipStream_t stream) {
  (void)in_sizes; (void)n_in; (void)out_size; (void)ws_size;
  const int B = 2, S = 2048, d = 1024, H = 16;
  const int M = B * S;   // 4096
  const int d4 = 4 * d;  // 4096

  const float* embed = (const float*)d_in[0];
  const float* Wk    = (const float*)d_in[1];
  const float* Wq    = (const float*)d_in[2];
  const float* Wv    = (const float*)d_in[3];
  const float* w1w   = (const float*)d_in[4];
  const float* w1b   = (const float*)d_in[5];
  const float* w2w   = (const float*)d_in[6];
  const float* w2b   = (const float*)d_in[7];
  const float* ln_g  = (const float*)d_in[8];
  const float* ln_b  = (const float*)d_in[9];
  const float* ln2_g = (const float*)d_in[10];
  const float* ln2_b = (const float*)d_in[11];

  char* w = (char*)d_ws;
  size_t off = 0;
  auto alloc = [&](size_t bytes) {
    char* p = w + off;
    off = (off + bytes + 255) & ~(size_t)255;
    return p;
  };
  u16* embed_bf = (u16*)alloc((size_t)M * d * 2);
  u16* wk_bf    = (u16*)alloc((size_t)d * d * 2);
  u16* wq_bf    = (u16*)alloc((size_t)d * d * 2);
  u16* wv_bf    = (u16*)alloc((size_t)d * d * 2);
  u16* w1_bf    = (u16*)alloc((size_t)d4 * d * 2);
  u16* w2_bf    = (u16*)alloc((size_t)d * d4 * 2);
  u16* Qbf      = (u16*)alloc((size_t)M * d * 2);  // |
  u16* Kbf      = (u16*)alloc((size_t)M * d * 2);  // | contiguous 32 MB,
  u16* Vbf      = (u16*)alloc((size_t)M * d * 2);  // | reused as `hidden`
  u16* Vt       = (u16*)alloc((size_t)M * d * 2);  // |
  float* attnf  = (float*)alloc((size_t)M * d * 4);  // reused as out2
  float* xf     = (float*)alloc((size_t)M * d * 4);
  u16* xbf      = (u16*)alloc((size_t)M * d * 2);
  u16* hidden   = Qbf;     // M x 4d bf16 aliases Q/K/V/Vt (dead after attention)
  float* out2   = attnf;   // dead after LN1

  auto cvt = [&](const float* src, u16* dst, size_t n) {
    cvt_f32_to_bf16<<<(unsigned)((n + 255) / 256), 256, 0, stream>>>(src, dst, n);
  };
  cvt(embed, embed_bf, (size_t)M * d);
  cvt(Wk, wk_bf, (size_t)d * d);
  cvt(Wq, wq_bf, (size_t)d * d);
  cvt(Wv, wv_bf, (size_t)d * d);
  cvt(w1w, w1_bf, (size_t)d4 * d);
  cvt(w2w, w2_bf, (size_t)d * d4);

  dim3 g_qkv(M / 128, d / 128);
  gemm_nt_bf16<4><<<g_qkv, 256, 0, stream>>>(embed_bf, wq_bf, nullptr, Qbf, M, d, d);
  gemm_nt_bf16<0><<<g_qkv, 256, 0, stream>>>(embed_bf, wk_bf, nullptr, Kbf, M, d, d);
  gemm_nt_bf16<0><<<g_qkv, 256, 0, stream>>>(embed_bf, wv_bf, nullptr, Vbf, M, d, d);

  {
    size_t n = (size_t)M * d;
    transpose_v_kernel<<<(unsigned)((n + 255) / 256), 256, 0, stream>>>(Vbf, Vt, S, d,
                                                                        H, n);
  }

  attn_kernel<<<dim3(S / 64, B * H), 128, 0, stream>>>(Qbf, Kbf, Vt, attnf, S, d, H);

  ln_kernel<<<M, 256, 0, stream>>>(attnf, embed, ln_g, ln_b, xf, xbf, d);

  gemm_nt_bf16<2><<<dim3(M / 128, d4 / 128), 256, 0, stream>>>(xbf, w1_bf, w1b,
                                                               hidden, M, d4, d);
  gemm_nt_bf16<3><<<dim3(M / 128, d / 128), 256, 0, stream>>>(hidden, w2_bf, w2b,
                                                              out2, M, d, d4);

  ln_kernel<<<M, 256, 0, stream>>>(out2, xf, ln2_g, ln2_b, (float*)d_out, nullptr, d);
}
